// NetEDU_46866683133954
// MI455X (gfx1250) — matrix operand order repaired and checked by reading, not compiled, not run on hardware
//
#include <hip/hip_runtime.h>
#include <hip/hip_bf16.h>
#include <math.h>

// ---------------- model constants ----------------
static constexpr int LYR = 12, H = 768, NH = 12, DH = 64, FFD = 3072;
static constexpr int BB = 8, TT = 512, KTAG = 7, TAG_START = 5;
static constexpr float NEGV = -10000.0f;

typedef __bf16 bf16;
typedef __attribute__((ext_vector_type(16))) __bf16 v16bf;
typedef __attribute__((ext_vector_type(8)))  float  v8f;
typedef __attribute__((ext_vector_type(4)))  int    v4i;

// LDS row stride (elements): 32 used + 8 pad -> 80B rows, keeps 16B alignment
static constexpr int LP = 40;

enum { EPI_LIN_F32 = 0, EPI_QK = 1, EPI_VT = 2, EPI_CTX = 3, EPI_S = 4, EPI_FF1 = 5 };

// CDNA5 async memory->LDS staging (ASYNCcnt) if the toolchain declares it
#if defined(__AMDGCN__) && __has_builtin(__builtin_amdgcn_global_load_async_to_lds_b128) && \
    __has_builtin(__builtin_amdgcn_s_wait_asynccnt)
#define USE_ASYNC_LDS 1
typedef __attribute__((address_space(1))) v4i g_v4i;   // global (prints as __device__)
typedef __attribute__((address_space(3))) v4i l_v4i;   // LDS    (prints as __shared__)
#define STAGE16(gp, lp) __builtin_amdgcn_global_load_async_to_lds_b128( \
    (g_v4i*)(void*)(gp), (l_v4i*)(void*)(lp), 0, 0)
#else
#define USE_ASYNC_LDS 0
#define STAGE16(gp, lp) (*(uint4*)(lp) = *(const uint4*)(gp))
#endif

// ---------------- weight transpose + fp32 -> bf16 ----------------
// in: [z][R][C] f32 row-major   out: [z][C][R] bf16
__global__ __launch_bounds__(256) void k_transpose_bf16(
    const float* __restrict__ in, bf16* __restrict__ out, int R, int C) {
  __shared__ float tile[32][33];
  const long long zo = (long long)blockIdx.z * R * C;
  const int c0 = blockIdx.x * 32, r0 = blockIdx.y * 32;
  const int tx = threadIdx.x & 31, ty = threadIdx.x >> 5;   // 8 rows of 32
  for (int i = ty; i < 32; i += 8)
    tile[i][tx] = in[zo + (long long)(r0 + i) * C + (c0 + tx)];
  __syncthreads();
  for (int i = ty; i < 32; i += 8)
    out[zo + (long long)(c0 + i) * R + (r0 + tx)] = (bf16)tile[tx][i];
}

// ---------------- generic bf16 WMMA GEMM:  C = A[M,K] * Bt[N,K]^T ----------------
// 128 threads (4 waves). Block tile 128(M) x 64(N); each wave owns 64x32
// (4x2 WMMA tiles -> 8 v_wmma per K-step). K-step = 32, LDS double-buffered,
// one barrier per K-step, staging via async-to-LDS (uniform control flow).
template <int EPI>
__global__ __launch_bounds__(128) void k_gemm_bf16(
    const bf16* __restrict__ A, const bf16* __restrict__ Bt,
    const float* __restrict__ bias, float* __restrict__ outF,
    bf16* __restrict__ outB, int M, int N, int Kd,
    long long sA, long long sB, long long sO) {
  __shared__ bf16 At[2][128 * LP];
  __shared__ bf16 Bl[2][64 * LP];
  const int bz = blockIdx.z;
  A  += (long long)bz * sA;
  Bt += (long long)bz * sB;

  const int tid  = threadIdx.x;
  const int lane = tid & 31;
  const int wave = tid >> 5;
  const int wm = (wave >> 1) * 64;       // wave row offset within 128
  const int wn = (wave & 1)  * 32;       // wave col offset within 64
  const int m0 = blockIdx.y * 128, n0 = blockIdx.x * 64;

  v8f acc[4][2] = {};

  const int fr = lane & 15;              // fragment row/col within 16
  const int kh = lane >> 4;              // K-half selector

  // staging map (uniform per thread, no divergence):
  //   chunk i=0..3 -> A row (tid>>2)+32i ; chunk i=0..1 -> B row (tid>>2)+32i
  //   16B chunk at element offset (tid&3)*8 within the 32-element K-row
  const int srow = tid >> 2;             // 0..31
  const int sch  = (tid & 3) << 3;       // 0/8/16/24
  const bf16* gA = A  + (long long)(m0 + srow) * Kd + sch;
  const bf16* gB = Bt + (long long)(n0 + srow) * Kd + sch;

  auto stage = [&](int k0, int buf) {
#pragma unroll
    for (int i = 0; i < 4; ++i) {        // A rows (always < 128)
      const int row = srow + 32 * i;
      STAGE16(gA + (long long)(32 * i) * Kd + k0, &At[buf][row * LP + sch]);
    }
#pragma unroll
    for (int i = 0; i < 2; ++i) {        // B rows (always < 64)
      const int r2 = srow + 32 * i;
      STAGE16(gB + (long long)(32 * i) * Kd + k0, &Bl[buf][r2 * LP + sch]);
    }
  };

  stage(0, 0);
#if USE_ASYNC_LDS
  __builtin_amdgcn_s_wait_asynccnt(0);
#endif
  __syncthreads();

  const int nK = Kd >> 5;
  int cur = 0;
  for (int ks = 0; ks < nK; ++ks) {
    if (ks + 1 < nK) stage((ks + 1) << 5, cur ^ 1);  // overlap with WMMAs below

    union Frag { uint4 u[2]; v16bf v; };
    Frag af[4], bfr[2];
    // A 16x32 bf16: lanes0-15 hold K0..7 & K16..23 of row fr; lanes16-31 K8..15 & K24..31
#pragma unroll
    for (int i = 0; i < 4; ++i) {
      const bf16* p = &At[cur][(wm + i * 16 + fr) * LP + kh * 8];
      af[i].u[0] = *(const uint4*)p;
      af[i].u[1] = *(const uint4*)(p + 16);
    }
    // B 32x16 bf16: lanes0-15 hold K0..7 & K16..23 of col fr; lanes16-31 K8..15 & K24..31
#pragma unroll
    for (int j = 0; j < 2; ++j) {
      const bf16* p = &Bl[cur][(wn + j * 16 + fr) * LP + kh * 8];
      bfr[j].u[0] = *(const uint4*)p;
      bfr[j].u[1] = *(const uint4*)(p + 16);
    }
#pragma unroll
    for (int i = 0; i < 4; ++i)
#pragma unroll
      for (int j = 0; j < 2; ++j)
        acc[i][j] = __builtin_amdgcn_wmma_f32_16x16x32_bf16(
            false, af[i].v, false, bfr[j].v, (short)0, acc[i][j], false, false);

#if USE_ASYNC_LDS
    __builtin_amdgcn_s_wait_asynccnt(0);
#endif
    __syncthreads();
    cur ^= 1;
  }

  // epilogue: C 16x16 f32 layout -> lane holds N=lane&15, M=(lane>>4)*8 + r
  const int cn = lane & 15;
  const int cm = (lane >> 4) * 8;
#pragma unroll
  for (int i = 0; i < 4; ++i)
#pragma unroll
    for (int j = 0; j < 2; ++j)
#pragma unroll
      for (int r = 0; r < 8; ++r) {
        const int gm = m0 + wm + i * 16 + cm + r;
        const int gn = n0 + wn + j * 16 + cn;
        float v = acc[i][j][r];
        if constexpr (EPI == EPI_LIN_F32) {
          outF[(long long)gm * N + gn] = v + bias[gn];
        } else if constexpr (EPI == EPI_QK) {
          const int b = gm / TT, t = gm % TT, hh = gn / DH, d = gn % DH;
          outB[(((long long)(b * NH + hh)) * TT + t) * DH + d] = (bf16)(v + bias[gn]);
        } else if constexpr (EPI == EPI_VT) {
          const int b = gm / TT, t = gm % TT, hh = gn / DH, d = gn % DH;
          outB[(((long long)(b * NH + hh)) * DH + d) * TT + t] = (bf16)(v + bias[gn]);
        } else if constexpr (EPI == EPI_CTX) {
          const int b = bz / NH, hh = bz % NH;
          outB[((long long)(b * TT + gm)) * H + hh * DH + gn] = (bf16)v;
        } else if constexpr (EPI == EPI_S) {
          outF[(long long)bz * sO + (long long)gm * N + gn] = v;
        } else if constexpr (EPI == EPI_FF1) {
          float x = v + bias[gn];
          float g = 0.5f * x * (1.0f + erff(x * 0.70710678118654752f)); // exact gelu
          outB[(long long)gm * N + gn] = (bf16)g;
        }
      }
}

// ---------------- embedding + LayerNorm ----------------
__global__ __launch_bounds__(256) void k_embed_ln(
    const int* __restrict__ X, const float* __restrict__ we,
    const float* __restrict__ pe, const float* __restrict__ te,
    const float* __restrict__ g, const float* __restrict__ b,
    float* __restrict__ outF, bf16* __restrict__ outB) {
  __shared__ float buf[H];
  __shared__ float r1[8], r2[8];
  const int tok = blockIdx.x, t = tok % TT;
  const int w = X[tok];
  float s = 0.f, q = 0.f;
  for (int i = threadIdx.x; i < H; i += 256) {
    float v = we[(long long)w * H + i] + pe[(long long)t * H + i] + te[i];
    buf[i] = v; s += v; q += v * v;
  }
  for (int off = 16; off; off >>= 1) { s += __shfl_xor(s, off, 32); q += __shfl_xor(q, off, 32); }
  if ((threadIdx.x & 31) == 0) { r1[threadIdx.x >> 5] = s; r2[threadIdx.x >> 5] = q; }
  __syncthreads();
  if (threadIdx.x == 0) {
    float ss = 0.f, qq = 0.f;
    for (int i = 0; i < 8; ++i) { ss += r1[i]; qq += r2[i]; }
    float m = ss / H;
    r1[0] = m; r2[0] = rsqrtf(qq / H - m * m + 1e-12f);
  }
  __syncthreads();
  const float m = r1[0], inv = r2[0];
  for (int i = threadIdx.x; i < H; i += 256) {
    float y = (buf[i] - m) * inv * g[i] + b[i];
    outF[(long long)tok * H + i] = y;
    outB[(long long)tok * H + i] = (bf16)y;
  }
}

// ---------------- residual add + LayerNorm ----------------
__global__ __launch_bounds__(256) void k_add_ln(
    const float* __restrict__ base, const float* __restrict__ add,
    const float* __restrict__ g, const float* __restrict__ b,
    float* __restrict__ outF, bf16* __restrict__ outB) {
  __shared__ float buf[H];
  __shared__ float r1[8], r2[8];
  const int tok = blockIdx.x;
  float s = 0.f, q = 0.f;
  for (int i = threadIdx.x; i < H; i += 256) {
    float v = base[(long long)tok * H + i] + add[(long long)tok * H + i];
    buf[i] = v; s += v; q += v * v;
  }
  for (int off = 16; off; off >>= 1) { s += __shfl_xor(s, off, 32); q += __shfl_xor(q, off, 32); }
  if ((threadIdx.x & 31) == 0) { r1[threadIdx.x >> 5] = s; r2[threadIdx.x >> 5] = q; }
  __syncthreads();
  if (threadIdx.x == 0) {
    float ss = 0.f, qq = 0.f;
    for (int i = 0; i < 8; ++i) { ss += r1[i]; qq += r2[i]; }
    float m = ss / H;
    r1[0] = m; r2[0] = rsqrtf(qq / H - m * m + 1e-12f);
  }
  __syncthreads();
  const float m = r1[0], inv = r2[0];
  for (int i = threadIdx.x; i < H; i += 256) {
    float y = (buf[i] - m) * inv * g[i] + b[i];
    outF[(long long)tok * H + i] = y;
    outB[(long long)tok * H + i] = (bf16)y;
  }
}

// ---------------- softmax over rows of 512 (scale folded), f32 -> bf16 ----------------
__global__ __launch_bounds__(256) void k_softmax(
    const float* __restrict__ S, bf16* __restrict__ P) {
  const int row = blockIdx.x * 8 + (threadIdx.x >> 5);
  const int lane = threadIdx.x & 31;
  const float* src = S + (long long)row * TT;
  float v[16], mx = -1e30f;
  for (int i = 0; i < 16; ++i) { v[i] = src[lane + i * 32] * 0.125f; mx = fmaxf(mx, v[i]); }
  for (int off = 16; off; off >>= 1) mx = fmaxf(mx, __shfl_xor(mx, off, 32));
  float sum = 0.f;
  for (int i = 0; i < 16; ++i) { v[i] = __expf(v[i] - mx); sum += v[i]; }
  for (int off = 16; off; off >>= 1) sum += __shfl_xor(sum, off, 32);
  const float inv = 1.f / sum;
  bf16* dst = P + (long long)row * TT;
  for (int i = 0; i < 16; ++i) dst[lane + i * 32] = (bf16)(v[i] * inv);
}

// ---------------- tag projection: feats = h @ W_tag + b_tag ----------------
__global__ __launch_bounds__(256) void k_tagproj(
    const float* __restrict__ h, const float* __restrict__ Wt,
    const float* __restrict__ bt, float* __restrict__ feats) {
  const int idx = blockIdx.x * 256 + threadIdx.x;
  if (idx >= BB * TT * KTAG) return;
  const int tok = idx / KTAG, kk = idx % KTAG;
  float s = bt[kk];
  const float* hr = h + (long long)tok * H;
  for (int i = 0; i < H; ++i) s += hr[i] * Wt[i * KTAG + kk];
  feats[idx] = s;
}

// ---------------- Viterbi decode (B=8, K=7) ----------------
__global__ __launch_bounds__(64) void k_viterbi(
    const float* __restrict__ feats, const float* __restrict__ trans,
    float* __restrict__ path, int* __restrict__ bptr) {
  __shared__ float sc[BB][KTAG], nsc[BB][KTAG];
  __shared__ float tr[KTAG][KTAG];
  const int tid = threadIdx.x;
  if (tid < KTAG * KTAG) tr[tid / KTAG][tid % KTAG] = trans[tid];
  const int b = tid / KTAG, k = tid % KTAG;
  const bool act = tid < BB * KTAG;
  if (act) sc[b][k] = (k == TAG_START) ? 0.f : NEGV;
  __syncthreads();
  for (int t = 0; t < TT; ++t) {
    if (act) {
      float best = -1e30f; int ba = 0;
      for (int p = 0; p < KTAG; ++p) {
        float c = sc[b][p] + tr[k][p];
        if (c > best) { best = c; ba = p; }
      }
      nsc[b][k] = best + feats[((long long)b * TT + t) * KTAG + k];
      bptr[(t * BB + b) * KTAG + k] = ba;
    }
    __syncthreads();
    if (act) sc[b][k] = nsc[b][k];
    __syncthreads();
  }
  if (tid < BB) {
    float best = -1e30f; int tag = 0;
    for (int kk = 0; kk < KTAG; ++kk)
      if (sc[tid][kk] > best) { best = sc[tid][kk]; tag = kk; }
    path[(long long)tid * TT + (TT - 1)] = (float)tag;
    for (int t = TT - 1; t > 0; --t) {
      tag = bptr[(t * BB + tid) * KTAG + tag];
      path[(long long)tid * TT + (t - 1)] = (float)tag;
    }
  }
}

// ---------------- host launch ----------------
extern "C" void kernel_launch(void* const* d_in, const int* in_sizes, int n_in,
                              void* d_out, int out_size, void* d_ws, size_t ws_size,
                              hipStream_t stream) {
  const int*   X        = (const int*)  d_in[0];
  const float* word_emb = (const float*)d_in[1];
  const float* pos_emb  = (const float*)d_in[2];
  const float* type_emb = (const float*)d_in[3];
  const float* emb_g    = (const float*)d_in[4];
  const float* emb_b    = (const float*)d_in[5];
  const float* Wq = (const float*)d_in[6];  const float* bq = (const float*)d_in[7];
  const float* Wk = (const float*)d_in[8];  const float* bk = (const float*)d_in[9];
  const float* Wv = (const float*)d_in[10]; const float* bv = (const float*)d_in[11];
  const float* Wo = (const float*)d_in[12]; const float* bo = (const float*)d_in[13];
  const float* ln1_g = (const float*)d_in[14]; const float* ln1_b = (const float*)d_in[15];
  const float* W1 = (const float*)d_in[16]; const float* b1 = (const float*)d_in[17];
  const float* W2 = (const float*)d_in[18]; const float* b2 = (const float*)d_in[19];
  const float* ln2_g = (const float*)d_in[20]; const float* ln2_b = (const float*)d_in[21];
  const float* W_tag = (const float*)d_in[22]; const float* b_tag = (const float*)d_in[23];
  const float* trans = (const float*)d_in[24];

  const int MTOK = BB * TT;          // 4096 tokens
  // ---- carve workspace ----
  char* base = (char*)d_ws; size_t off = 0;
  auto carve = [&](size_t bytes) -> void* {
    void* p = base + off; off = (off + bytes + 255) & ~(size_t)255; return p;
  };
  bf16* wqT = (bf16*)carve((size_t)LYR * H * H * 2);
  bf16* wkT = (bf16*)carve((size_t)LYR * H * H * 2);
  bf16* wvT = (bf16*)carve((size_t)LYR * H * H * 2);
  bf16* woT = (bf16*)carve((size_t)LYR * H * H * 2);
  bf16* w1T = (bf16*)carve((size_t)LYR * H * FFD * 2);
  bf16* w2T = (bf16*)carve((size_t)LYR * H * FFD * 2);
  float* hF  = (float*)carve((size_t)MTOK * H * 4);
  bf16*  hB  = (bf16*) carve((size_t)MTOK * H * 2);
  float* h2F = (float*)carve((size_t)MTOK * H * 4);
  bf16*  h2B = (bf16*) carve((size_t)MTOK * H * 2);
  bf16*  qb  = (bf16*) carve((size_t)MTOK * H * 2);
  bf16*  kb  = (bf16*) carve((size_t)MTOK * H * 2);
  bf16*  vtb = (bf16*) carve((size_t)MTOK * H * 2);
  bf16*  ctxb= (bf16*) carve((size_t)MTOK * H * 2);
  float* Sbuf= (float*)carve((size_t)BB * NH * TT * TT * 4);
  bf16*  Pbuf= (bf16*) carve((size_t)BB * NH * TT * TT * 2);
  bf16*  ff1b= (bf16*) carve((size_t)MTOK * FFD * 2);
  float* tmpF= (float*)carve((size_t)MTOK * H * 4);
  int*   bptr= (int*)  carve((size_t)TT * BB * KTAG * 4);

  // ---- weight prep: fp32 [R][C] -> bf16 [C][R], all layers at once (z = L) ----
  k_transpose_bf16<<<dim3(H / 32, H / 32, LYR), 256, 0, stream>>>(Wq, wqT, H, H);
  k_transpose_bf16<<<dim3(H / 32, H / 32, LYR), 256, 0, stream>>>(Wk, wkT, H, H);
  k_transpose_bf16<<<dim3(H / 32, H / 32, LYR), 256, 0, stream>>>(Wv, wvT, H, H);
  k_transpose_bf16<<<dim3(H / 32, H / 32, LYR), 256, 0, stream>>>(Wo, woT, H, H);
  k_transpose_bf16<<<dim3(FFD / 32, H / 32, LYR), 256, 0, stream>>>(W1, w1T, H, FFD);
  k_transpose_bf16<<<dim3(H / 32, FFD / 32, LYR), 256, 0, stream>>>(W2, w2T, FFD, H);

  // ---- embeddings + LN ----
  k_embed_ln<<<MTOK, 256, 0, stream>>>(X, word_emb, pos_emb, type_emb, emb_g, emb_b, hF, hB);

  const long long sQK = (long long)TT * DH;     // per-(b,h) stride for Q/K/Vt
  const long long sSS = (long long)TT * TT;

  for (int l = 0; l < LYR; ++l) {
    const bf16* wq_l = wqT + (size_t)l * H * H;
    const bf16* wk_l = wkT + (size_t)l * H * H;
    const bf16* wv_l = wvT + (size_t)l * H * H;
    const bf16* wo_l = woT + (size_t)l * H * H;
    const bf16* w1_l = w1T + (size_t)l * H * FFD;
    const bf16* w2_l = w2T + (size_t)l * H * FFD;

    // Q, K (head layout [B,NH,T,DH]), V transposed ([B,NH,DH,T])
    k_gemm_bf16<EPI_QK><<<dim3(H / 64, MTOK / 128, 1), 128, 0, stream>>>(
        hB, wq_l, bq + l * H, nullptr, qb, MTOK, H, H, 0, 0, 0);
    k_gemm_bf16<EPI_QK><<<dim3(H / 64, MTOK / 128, 1), 128, 0, stream>>>(
        hB, wk_l, bk + l * H, nullptr, kb, MTOK, H, H, 0, 0, 0);
    k_gemm_bf16<EPI_VT><<<dim3(H / 64, MTOK / 128, 1), 128, 0, stream>>>(
        hB, wv_l, bv + l * H, nullptr, vtb, MTOK, H, H, 0, 0, 0);

    // S = Q K^T per (b,h): batched z = 96
    k_gemm_bf16<EPI_S><<<dim3(TT / 64, TT / 128, BB * NH), 128, 0, stream>>>(
        qb, kb, nullptr, Sbuf, nullptr, TT, TT, DH, sQK, sQK, sSS);

    // softmax rows (scale 1/8 folded) -> bf16 P
    k_softmax<<<(BB * NH * TT) / 8, 256, 0, stream>>>(Sbuf, Pbuf);

    // ctx = P V  (Bt = V^T already) -> bf16 [B,T,H]
    k_gemm_bf16<EPI_CTX><<<dim3(DH / 64, TT / 128, BB * NH), 128, 0, stream>>>(
        Pbuf, vtb, nullptr, nullptr, ctxb, TT, DH, TT, sSS, sQK, 0);

    // O projection -> f32 tmp
    k_gemm_bf16<EPI_LIN_F32><<<dim3(H / 64, MTOK / 128, 1), 128, 0, stream>>>(
        ctxb, wo_l, bo + l * H, tmpF, nullptr, MTOK, H, H, 0, 0, 0);

    // h2 = LN(h + O)
    k_add_ln<<<MTOK, 256, 0, stream>>>(hF, tmpF, ln1_g + l * H, ln1_b + l * H, h2F, h2B);

    // FF1 with exact GELU -> bf16
    k_gemm_bf16<EPI_FF1><<<dim3(FFD / 64, MTOK / 128, 1), 128, 0, stream>>>(
        h2B, w1_l, b1 + l * FFD, nullptr, ff1b, MTOK, FFD, H, 0, 0, 0);

    // FF2 -> f32 tmp
    k_gemm_bf16<EPI_LIN_F32><<<dim3(H / 64, MTOK / 128, 1), 128, 0, stream>>>(
        ff1b, w2_l, b2 + l * H, tmpF, nullptr, MTOK, H, FFD, 0, 0, 0);

    // h = LN(h2 + FF2)
    k_add_ln<<<MTOK, 256, 0, stream>>>(h2F, tmpF, ln2_g + l * H, ln2_b + l * H, hF, hB);
  }

  // feats -> d_out[0 .. B*T*K)
  float* feats = (float*)d_out;
  k_tagproj<<<(MTOK * KTAG + 255) / 256, 256, 0, stream>>>(hF, W_tag, b_tag, feats);

  // Viterbi path -> d_out[B*T*K .. B*T*K + B*T)
  float* path = feats + (size_t)MTOK * KTAG;
  k_viterbi<<<1, 64, 0, stream>>>(feats, trans, path, bptr);
}
